// MAGNNModel_62414464745717
// MI455X (gfx1250) — matrix operand order repaired and checked by reading, not compiled, not run on hardware
//
#include <hip/hip_runtime.h>
#include <hip/hip_bf16.h>

typedef __attribute__((ext_vector_type(16))) _Float16 v16h;
typedef __attribute__((ext_vector_type(8)))  float    v8f;

#define D_    64
#define H_    4
#define AV_   128
#define RMAX_ 32
#define R20_  20
#define I5_   5
#define NREP_ (R20_*I5_)   /* 100 */
#define HD_   (H_*D_)      /* 256 */

// ---------------- kernel 0: weights f32 -> f16 ----------------
__global__ __launch_bounds__(256)
void cvt_w(const float* __restrict__ wu, const float* __restrict__ wr,
           _Float16* __restrict__ wu16, _Float16* __restrict__ wr16) {
  int i = blockIdx.x * 256 + threadIdx.x;           // 0 .. 2*AV*HD-1
  if (i < AV_*HD_)      wu16[i] = (_Float16)wu[i];
  else                  wr16[i - AV_*HD_] = (_Float16)wr[i - AV_*HD_];
}

// ---------------- kernel 1: fused per-element intra attention ----------------
__global__ __launch_bounds__(128)
void magnn_intra(const int*   __restrict__ user_ids,
                 const int*   __restrict__ recipe_ids,
                 const int*   __restrict__ user2recipes,
                 const int*   __restrict__ recipe2ing,
                 const float* __restrict__ user_emb,
                 const float* __restrict__ recipe_emb,
                 const float* __restrict__ ing_emb,
                 const float* __restrict__ attn_UR,
                 const float* __restrict__ attn_URIR,
                 float* __restrict__ stacked,   // [B,2,256]
                 float* __restrict__ rirv)      // [B,256]
{
  __shared__ float s_uf[D_];
  __shared__ int   s_recs[RMAX_];
  __shared__ float s_repsUR[RMAX_*D_];
  __shared__ int   s_ing[NREP_];
  __shared__ float s_repsI[NREP_*D_];
  __shared__ float s_e[NREP_*H_];      // reused: UR (128) then URIR (400)

  const int b = blockIdx.x, tid = threadIdx.x;
  const int u = user_ids[b];

  if (tid < D_)                      s_uf[tid]       = user_emb[(size_t)u*D_ + tid];
  if (tid >= D_ && tid < D_+RMAX_)   s_recs[tid-D_]  = user2recipes[(size_t)u*RMAX_ + (tid-D_)];
  __syncthreads();

  // UR reps = 0.5*(uf + rf)
  for (int idx = tid; idx < RMAX_*D_; idx += 128) {
    int n = idx >> 6, d = idx & 63;
    float rv = recipe_emb[(size_t)s_recs[n]*D_ + d];
    s_repsUR[idx] = 0.5f*(s_uf[d] + rv);
  }
  __syncthreads();

  // e_UR: one thread per (n,h), 32*4 = 128
  {
    int n = tid >> 2, h = tid & 3;
    const float* aw = attn_UR + h*(2*D_);
    float acc = 0.f;
    for (int d = 0; d < D_; ++d)
      acc += aw[d]*s_uf[d] + aw[D_+d]*s_repsUR[n*D_ + d];
    s_e[n*H_ + h] = (acc > 0.f) ? acc : 0.2f*acc;       // leaky_relu(0.2)
  }
  __syncthreads();
  if (tid < H_) {                                        // softmax over n per head
    float m = -1e30f;
    for (int n = 0; n < RMAX_; ++n) m = fmaxf(m, s_e[n*H_+tid]);
    float s = 0.f;
    for (int n = 0; n < RMAX_; ++n) s += __expf(s_e[n*H_+tid]-m);
    float inv = 1.f/s;
    for (int n = 0; n < RMAX_; ++n) s_e[n*H_+tid] = __expf(s_e[n*H_+tid]-m)*inv;
  }
  __syncthreads();
  for (int o = tid; o < HD_; o += 128) {                 // UR_out
    int h = o >> 6, d = o & 63;
    float acc = 0.f;
    for (int n = 0; n < RMAX_; ++n) acc += s_e[n*H_+h]*s_repsUR[n*D_+d];
    stacked[(size_t)b*2*HD_ + o] = acc;
  }
  __syncthreads();

  // URIR reps: 0.25*uf + 0.5*rf20 + 0.25*ing == repsUR[n/5] - 0.25*uf + 0.25*ing
  if (tid < NREP_) s_ing[tid] = recipe2ing[(size_t)s_recs[tid/I5_]*I5_ + (tid % I5_)];
  __syncthreads();
  for (int idx = tid; idx < NREP_*D_; idx += 128) {
    int n = idx >> 6, d = idx & 63;
    float iv = ing_emb[(size_t)s_ing[n]*D_ + d];
    s_repsI[idx] = s_repsUR[(n/I5_)*D_ + d] - 0.25f*s_uf[d] + 0.25f*iv;
  }
  __syncthreads();
  for (int j = tid; j < NREP_*H_; j += 128) {            // e_URIR (400 dots)
    int n = j >> 2, h = j & 3;
    const float* aw = attn_URIR + h*(2*D_);
    float acc = 0.f;
    for (int d = 0; d < D_; ++d)
      acc += aw[d]*s_uf[d] + aw[D_+d]*s_repsI[n*D_+d];
    s_e[n*H_+h] = (acc > 0.f) ? acc : 0.2f*acc;
  }
  __syncthreads();
  if (tid < H_) {
    float m = -1e30f;
    for (int n = 0; n < NREP_; ++n) m = fmaxf(m, s_e[n*H_+tid]);
    float s = 0.f;
    for (int n = 0; n < NREP_; ++n) s += __expf(s_e[n*H_+tid]-m);
    float inv = 1.f/s;
    for (int n = 0; n < NREP_; ++n) s_e[n*H_+tid] = __expf(s_e[n*H_+tid]-m)*inv;
  }
  __syncthreads();
  for (int o = tid; o < HD_; o += 128) {                 // URIR_out
    int h = o >> 6, d = o & 63;
    float acc = 0.f;
    for (int n = 0; n < NREP_; ++n) acc += s_e[n*H_+h]*s_repsI[n*D_+d];
    stacked[(size_t)b*2*HD_ + HD_ + o] = acc;
  }

  // RIR_out = (2/3)*rfeat tiled over heads (uniform softmax; attn_RIR drops out)
  const int r = recipe_ids[b];
  for (int o = tid; o < HD_; o += 128)
    rirv[(size_t)b*HD_ + o] = (2.f/3.f) * recipe_emb[(size_t)r*D_ + (o & 63)];
}

// ---------------- kernel 2: out = tanh(A[M,256] @ W[128,256]^T + bias) via WMMA ----------------
__global__ __launch_bounds__(256)
void gemm_tanh_wmma(const float*    __restrict__ A,     // [M, 256] f32 activations
                    const _Float16* __restrict__ W,     // [128, 256] f16 weights
                    const float*    __restrict__ bias,  // [128]
                    float*          __restrict__ out)   // [M, 128] f32
{
  const int wave = threadIdx.x >> 5;        // N tile (0..7)
  const int lane = threadIdx.x & 31;
  const int lr   = lane & 15;
  const int lhi  = lane >> 4;               // 0/1
  const int row  = blockIdx.x*16 + lr;      // A row this lane carries
  const int ncol = wave*16 + lr;            // output column / W row this lane carries
  const float*    arow = A + (size_t)row*HD_;
  const _Float16* wrow = W + (size_t)ncol*HD_;

  v8f c = {};
  #pragma unroll
  for (int k0 = 0; k0 < HD_; k0 += 32) {
    // A fragment (ISA 16-bit A 16x32): halves k0+lhi*8+[0..7] and k0+16+lhi*8+[0..7]
    const float4* ap = (const float4*)(arow + k0 + lhi*8);
    float4 a0 = ap[0], a1 = ap[1], a2 = ap[4], a3 = ap[5];
    v16h a;
    a[0]=(_Float16)a0.x;  a[1]=(_Float16)a0.y;  a[2]=(_Float16)a0.z;  a[3]=(_Float16)a0.w;
    a[4]=(_Float16)a1.x;  a[5]=(_Float16)a1.y;  a[6]=(_Float16)a1.z;  a[7]=(_Float16)a1.w;
    a[8]=(_Float16)a2.x;  a[9]=(_Float16)a2.y;  a[10]=(_Float16)a2.z; a[11]=(_Float16)a2.w;
    a[12]=(_Float16)a3.x; a[13]=(_Float16)a3.y; a[14]=(_Float16)a3.z; a[15]=(_Float16)a3.w;
    // B fragment (dense 32x16): column n=lr holds k = k0+lhi*8+[0..7] and k0+16+lhi*8+[0..7] (matches A)
    v16h blo = *(const v16h*)(wrow + k0), bhi = *(const v16h*)(wrow + k0 + 16);
    v16h bfrag = lhi ? __builtin_shufflevector(blo, bhi, 8,9,10,11,12,13,14,15,24,25,26,27,28,29,30,31)
                     : __builtin_shufflevector(blo, bhi, 0,1,2,3,4,5,6,7,16,17,18,19,20,21,22,23);
    c = __builtin_amdgcn_wmma_f32_16x16x32_f16(false, a, false, bfrag,
                                               (short)0, c, false, false);
  }
  const float bv = bias[ncol];
  #pragma unroll
  for (int r = 0; r < 8; ++r) {             // C/D: M = r + 8*lhi, N = lr
    int m = blockIdx.x*16 + r + lhi*8;
    out[(size_t)m*AV_ + wave*16 + lr] = tanhf(c[r] + bv);
  }
}

// ---------------- kernel 3: inter-attention + final dot ----------------
__device__ __forceinline__ float blk_reduce128(float v, float* s) {
  int tid = threadIdx.x;
  s[tid] = v; __syncthreads();
  for (int off = 64; off >= 1; off >>= 1) {
    if (tid < off) s[tid] += s[tid + off];
    __syncthreads();
  }
  float r = s[0]; __syncthreads();
  return r;
}

__global__ __launch_bounds__(128)
void magnn_final(const float* __restrict__ whU,     // [B,2,128]
                 const float* __restrict__ whR,     // [B,128]
                 const float* __restrict__ stacked, // [B,2,256]
                 const float* __restrict__ rirv,    // [B,256]
                 const float* __restrict__ q_u,
                 const float* __restrict__ b_r,
                 const float* __restrict__ q_r,
                 float* __restrict__ outp)
{
  __shared__ float s_red[128];
  __shared__ float s_coef[3];
  const int b = blockIdx.x, tid = threadIdx.x;
  const float qu = q_u[tid], qr = q_r[tid];

  float su0 = blk_reduce128(whU[((size_t)b*2+0)*AV_ + tid]*qu, s_red);
  float su1 = blk_reduce128(whU[((size_t)b*2+1)*AV_ + tid]*qu, s_red);
  float sr0 = blk_reduce128(tanhf(b_r[tid])*qr, s_red);              // RU_out == 0 path
  float sr1 = blk_reduce128(whR[(size_t)b*AV_ + tid]*qr, s_red);

  if (tid == 0) {
    float m  = fmaxf(su0, su1);
    float e0 = __expf(su0-m), e1 = __expf(su1-m);
    s_coef[0] = e0/(e0+e1);  s_coef[1] = e1/(e0+e1);
    float mr = fmaxf(sr0, sr1);
    float f0 = __expf(sr0-mr), f1 = __expf(sr1-mr);
    s_coef[2] = f1/(f0+f1);                                          // hr = ar1 * RIR_out
  }
  __syncthreads();
  const float au0 = s_coef[0], au1 = s_coef[1], ar1 = s_coef[2];
  const float* st = stacked + (size_t)b*2*HD_;
  const float* rv = rirv   + (size_t)b*HD_;
  float p = 0.f;
  for (int o = tid; o < HD_; o += 128) {
    float hu = au0*st[o] + au1*st[HD_+o];
    p += hu * rv[o];
  }
  float dot = blk_reduce128(p, s_red);
  if (tid == 0) outp[b] = ar1 * dot;
}

// ---------------- launcher ----------------
extern "C" void kernel_launch(void* const* d_in, const int* in_sizes, int n_in,
                              void* d_out, int out_size, void* d_ws, size_t ws_size,
                              hipStream_t stream) {
  const int*   user_ids     = (const int*)d_in[0];
  const int*   recipe_ids   = (const int*)d_in[1];
  const int*   user2recipes = (const int*)d_in[2];
  const int*   recipe2ing   = (const int*)d_in[3];
  const float* user_emb     = (const float*)d_in[4];
  const float* recipe_emb   = (const float*)d_in[5];
  const float* ing_emb      = (const float*)d_in[6];
  const float* attn_UR      = (const float*)d_in[7];
  const float* attn_URIR    = (const float*)d_in[8];
  /* d_in[9] attn_RIR unused: uniform softmax over identical reps */
  const float* W_u = (const float*)d_in[10];
  const float* b_u = (const float*)d_in[11];
  const float* q_u = (const float*)d_in[12];
  const float* W_r = (const float*)d_in[13];
  const float* b_r = (const float*)d_in[14];
  const float* q_r = (const float*)d_in[15];
  (void)n_in; (void)out_size; (void)ws_size;

  const int B = in_sizes[0];

  char* ws = (char*)d_ws;
  size_t off = 0;
  auto wsalloc = [&](size_t bytes) -> void* {
    void* p = ws + off;
    off = (off + bytes + 255) & ~(size_t)255;
    return p;
  };
  float*    stacked = (float*)   wsalloc((size_t)B*2*HD_*sizeof(float)); // 8 MB
  float*    rirv    = (float*)   wsalloc((size_t)B*HD_*sizeof(float));   // 4 MB
  float*    whU     = (float*)   wsalloc((size_t)B*2*AV_*sizeof(float)); // 4 MB
  float*    whR     = (float*)   wsalloc((size_t)B*AV_*sizeof(float));   // 2 MB
  _Float16* wu16    = (_Float16*)wsalloc((size_t)AV_*HD_*sizeof(_Float16));
  _Float16* wr16    = (_Float16*)wsalloc((size_t)AV_*HD_*sizeof(_Float16));

  cvt_w<<<(2*AV_*HD_)/256, 256, 0, stream>>>(W_u, W_r, wu16, wr16);

  magnn_intra<<<B, 128, 0, stream>>>(user_ids, recipe_ids, user2recipes, recipe2ing,
                                     user_emb, recipe_emb, ing_emb,
                                     attn_UR, attn_URIR, stacked, rirv);

  gemm_tanh_wmma<<<(2*B)/16, 256, 0, stream>>>(stacked, wu16, b_u, whU);
  gemm_tanh_wmma<<<B/16,     256, 0, stream>>>(rirv,    wr16, b_r, whR);

  magnn_final<<<B, 128, 0, stream>>>(whU, whR, stacked, rirv, q_u, b_r, q_r, (float*)d_out);
}
